// GAT_11742440587855
// MI455X (gfx1250) — matrix operand order repaired and checked by reading, not compiled, not run on hardware
//
#include <hip/hip_runtime.h>
#include <hip/hip_bf16.h>

// ---------------------------------------------------------------------------
// 3-layer GAT for gfx1250 (MI455X). GEMMs via v_wmma_f32_16x16x32_bf16 with
// fragment-swizzled LDS (vector ds_load_b128 feeds) and fully compile-time,
// wave-uniform tile mapping (EXEC all-ones through the WMMA pipeline).
// Segment softmax / aggregation via native global float atomics.
// ---------------------------------------------------------------------------

typedef __bf16 bf16;
typedef __attribute__((ext_vector_type(16))) __bf16 v16bf;
typedef __attribute__((ext_vector_type(8)))  float  v8f;

#define N_NODES   50000
#define N_FEAT    128
#define N_EDGES   800000
#define E_TOT     (N_EDGES + N_NODES)   // with self loops
#define HEADS1    4
#define HID1      64
#define HC1       (HEADS1 * HID1)       // 256
#define HID2      128
#define N_CLASSES 10
#define N_GRAPHS  512
#define SLOPE_ATT 0.2f
#define SLOPE_ACT 0.01f

// ---------------- order-preserving float<->uint encoding for atomicMax -----
__device__ __forceinline__ unsigned fenc(float f) {
    unsigned u = __float_as_uint(f);
    return (u & 0x80000000u) ? ~u : (u | 0x80000000u);
}
__device__ __forceinline__ float fdec(unsigned u) {
    unsigned b = (u & 0x80000000u) ? (u & 0x7FFFFFFFu) : ~u;
    return __uint_as_float(b);
}
#define ENC_NEG_INF 0x007FFFFFu   // fenc(-inf)

__device__ __forceinline__ float leaky(float x, float s) { return x > 0.0f ? x : s * x; }

// ---------------- fills ----------------------------------------------------
__global__ void fill_f32(float* p, float v, int n) {
    int i = blockIdx.x * blockDim.x + threadIdx.x;
    if (i < n) p[i] = v;
}
__global__ void fill_u32(unsigned* p, unsigned v, int n) {
    int i = blockIdx.x * blockDim.x + threadIdx.x;
    if (i < n) p[i] = v;
}

// ---------------- WMMA GEMM: out[M,Nreal] = X[M,KC*32] @ W[KC*32,Nreal] ----
// Block: 128 threads (4 waves). LDS holds W and the current 16-row A tile
// PRE-SWIZZLED into per-lane fragment order, so each fragment load is one
// contiguous 32B read (2x ds_load_b128) per lane.
//
// A (16x32, 16-bit) lane layout: row = lane&15, kb = lane<16 ? 0 : 8,
//   element e -> K = kb + e (e<8) | kb + 8 + e (e>=8)
// B (32x16, 16-bit) lane layout: col = lane&15, kb = lane<16 ? 0 : 8,
//   element e -> K = kb + e (e<8) | kb + 8 + e (e>=8)
//
// Tile mapping is compile-time: NT>=4 -> wave owns tiles wave*TPW..+TPW-1
// (no guards, EXEC all-ones); NT==1 -> all waves compute tile 0, wave 0 stores.
#define GEMM_THREADS 128
template <int KC, int NT>
__global__ __launch_bounds__(GEMM_THREADS)
void gemm_wmma(const float* __restrict__ X, const float* __restrict__ W,
               float* __restrict__ out, int M, int Nreal) {
    constexpr int K    = KC * 32;
    constexpr int Npad = NT * 16;
    constexpr int TPW  = (NT + 3) / 4;   // tiles per wave: 4, 2, or 1

    __shared__ __align__(32) bf16 Bs[KC * NT * 512]; // [KC][NT][32 lanes][16]
    __shared__ __align__(32) bf16 As[KC * 512];      // [KC][32 lanes][16]

    int tid  = threadIdx.x;
    int lane = tid & 31;
    int wave = tid >> 5;

    // ---- stage W -> LDS in fragment-swizzled order -------------------------
    for (int idx = tid; idx < KC * NT * 512; idx += GEMM_THREADS) {
        int e   = idx & 15;
        int ln  = (idx >> 4) & 31;
        int nt  = (idx >> 9) % NT;
        int kc  = (idx >> 9) / NT;
        int col = ln & 15;
        int kb  = (ln < 16) ? 0 : 8;
        int kk  = kc * 32 + ((e < 8) ? (kb + e) : (kb + 8 + e));
        int n   = nt * 16 + col;
        Bs[idx] = (n < Nreal) ? (bf16)W[kk * Nreal + n] : (bf16)0.0f;
    }
    __syncthreads();

    int numMt = (M + 15) >> 4;

    for (int mt = blockIdx.x; mt < numMt; mt += gridDim.x) {
        // ---- stage A tile -> LDS in fragment-swizzled order ----------------
        for (int idx = tid; idx < KC * 512; idx += GEMM_THREADS) {
            int e   = idx & 15;
            int ln  = (idx >> 4) & 31;
            int kc  = idx >> 9;
            int row = ln & 15;
            int kb  = (ln < 16) ? 0 : 8;
            int kk  = (e < 8) ? (kb + e) : (kb + 8 + e);
            int g   = mt * 16 + row;
            As[idx] = (g < M) ? (bf16)X[(size_t)g * K + kc * 32 + kk] : (bf16)0.0f;
        }
        __syncthreads();

        v8f acc[TPW];
#pragma unroll
        for (int t = 0; t < TPW; ++t)
#pragma unroll
            for (int r = 0; r < 8; ++r) acc[t][r] = 0.0f;

#pragma unroll
        for (int kc = 0; kc < KC; ++kc) {
            v16bf a = *(const v16bf*)&As[(kc * 32 + lane) * 16];
#pragma unroll
            for (int t = 0; t < TPW; ++t) {
                int nt = (NT >= 4) ? (wave * TPW + t) : 0;   // uniform, no guard
                v16bf b = *(const v16bf*)&Bs[((kc * NT + nt) * 32 + lane) * 16];
                acc[t] = __builtin_amdgcn_wmma_f32_16x16x32_bf16(
                    false, a, false, b, (short)0, acc[t], false, false);
            }
        }

        // ---- store D: VGPR r -> row (lane<16 ? r : 8+r), col = lane&15 -----
        if (NT >= 4 || wave == 0) {
            int col = lane & 15;
            int mb  = (lane < 16) ? 0 : 8;
#pragma unroll
            for (int t = 0; t < TPW; ++t) {
                int nt = (NT >= 4) ? (wave * TPW + t) : 0;
                int n  = nt * 16 + col;
                if (n < Nreal) {
#pragma unroll
                    for (int r = 0; r < 8; ++r) {
                        int g = mt * 16 + mb + r;
                        if (g < M) out[(size_t)g * Nreal + n] = acc[t][r];
                    }
                }
            }
        }
        __syncthreads();
    }
}

// ---------------- attention coefficients: als/ald[n,h] = <h[n,h,:], a[h,:]> -
__global__ void attn_coef(const float* __restrict__ hmat,
                          const float* __restrict__ asrc,
                          const float* __restrict__ adst,
                          float* __restrict__ als, float* __restrict__ ald,
                          int M, int H, int C) {
    int i = blockIdx.x * blockDim.x + threadIdx.x;
    if (i >= M * H) return;
    int n = i / H, h = i % H;
    const float* row = hmat + (size_t)n * H * C + h * C;
    float ss = 0.0f, sd = 0.0f;
    for (int c = 0; c < C; ++c) {
        float v = row[c];
        ss += v * asrc[h * C + c];
        sd += v * adst[h * C + c];
    }
    als[i] = ss;
    ald[i] = sd;
}

// ---------------- segment max of edge scores (encoded atomicMax) -----------
__global__ void edge_max_kernel(const int* __restrict__ ei,
                                const float* __restrict__ als,
                                const float* __restrict__ ald,
                                unsigned* __restrict__ m, int H) {
    int e = blockIdx.x * blockDim.x + threadIdx.x;
    if (e >= E_TOT) return;
    int s, d;
    if (e < N_EDGES) { s = ei[e]; d = ei[N_EDGES + e]; }
    else             { s = d = e - N_EDGES; }
    for (int h = 0; h < H; ++h) {
        float sc = leaky(als[s * H + h] + ald[d * H + h], SLOPE_ATT);
        atomicMax(&m[d * H + h], fenc(sc));
    }
}

// ---------------- exp(e - m[dst]) and segment sum --------------------------
__global__ void edge_exp_kernel(const int* __restrict__ ei,
                                const float* __restrict__ als,
                                const float* __restrict__ ald,
                                const unsigned* __restrict__ m,
                                float* __restrict__ eex,
                                float* __restrict__ den, int H) {
    int e = blockIdx.x * blockDim.x + threadIdx.x;
    if (e >= E_TOT) return;
    int s, d;
    if (e < N_EDGES) { s = ei[e]; d = ei[N_EDGES + e]; }
    else             { s = d = e - N_EDGES; }
    for (int h = 0; h < H; ++h) {
        float sc = leaky(als[s * H + h] + ald[d * H + h], SLOPE_ATT);
        float ex = __expf(sc - fdec(m[d * H + h]));
        eex[(size_t)e * H + h] = ex;
        atomicAdd(&den[d * H + h], ex);
    }
}

// ---------------- weighted scatter aggregation: one wave per edge ----------
__global__ void edge_aggregate(const int* __restrict__ ei,
                               const float* __restrict__ hin,
                               const float* __restrict__ eex,
                               const float* __restrict__ den,
                               float* __restrict__ out, int HC, int H) {
    int gid  = blockIdx.x * blockDim.x + threadIdx.x;
    int lane = gid & 31;
    int e    = gid >> 5;
    if (e >= E_TOT) return;
    int s, d;
    if (e < N_EDGES) { s = ei[e]; d = ei[N_EDGES + e]; }
    else             { s = d = e - N_EDGES; }
    int C = HC / H;
    for (int idx = lane; idx < HC; idx += 32) {
        int h = idx / C;
        float alpha = eex[(size_t)e * H + h] / (den[d * H + h] + 1e-16f);
        atomicAdd(&out[(size_t)d * HC + idx], hin[(size_t)s * HC + idx] * alpha);
    }
}

// ---------------- bias add + optional leaky relu ---------------------------
__global__ void bias_act(float* __restrict__ x, const float* __restrict__ b,
                         int M, int HC, float slope, int apply) {
    int i = blockIdx.x * blockDim.x + threadIdx.x;
    if (i >= M * HC) return;
    float v = x[i] + b[i % HC];
    if (apply) v = leaky(v, slope);
    x[i] = v;
}

// ---------------- global add pool ------------------------------------------
__global__ void pool_kernel(const float* __restrict__ h3,
                            const int* __restrict__ batch,
                            float* __restrict__ pooled) {
    int i = blockIdx.x * blockDim.x + threadIdx.x;
    if (i >= N_NODES * N_CLASSES) return;
    int n = i / N_CLASSES, c = i % N_CLASSES;
    atomicAdd(&pooled[batch[n] * N_CLASSES + c], h3[i]);
}

// ---------------- log softmax + tuple output -------------------------------
__global__ void lsm_kernel(const float* __restrict__ pooled, float* __restrict__ out) {
    int g = blockIdx.x * blockDim.x + threadIdx.x;
    if (g >= N_GRAPHS) return;
    float mx = -3.402823466e38f;
    for (int c = 0; c < N_CLASSES; ++c) mx = fmaxf(mx, pooled[g * N_CLASSES + c]);
    float ssum = 0.0f;
    for (int c = 0; c < N_CLASSES; ++c) ssum += __expf(pooled[g * N_CLASSES + c] - mx);
    float lse = mx + __logf(ssum);
    for (int c = 0; c < N_CLASSES; ++c) {
        float p = pooled[g * N_CLASSES + c];
        out[g * N_CLASSES + c] = p - lse;                            // log_softmax
        out[N_GRAPHS * N_CLASSES + g * N_CLASSES + c] = p;          // pooled
    }
}

// ---------------------------------------------------------------------------
// Host-side launch orchestration
// ---------------------------------------------------------------------------
static inline int cdiv(long long a, long long b) { return (int)((a + b - 1) / b); }

extern "C" void kernel_launch(void* const* d_in, const int* in_sizes, int n_in,
                              void* d_out, int out_size, void* d_ws, size_t ws_size,
                              hipStream_t stream) {
    const float* x     = (const float*)d_in[0];
    const int*   ei    = (const int*)  d_in[1];   // [2, N_EDGES]
    const int*   batch = (const int*)  d_in[3];
    const float* W1 = (const float*)d_in[4];
    const float* as1 = (const float*)d_in[5];
    const float* ad1 = (const float*)d_in[6];
    const float* b1  = (const float*)d_in[7];
    const float* W2 = (const float*)d_in[8];
    const float* as2 = (const float*)d_in[9];
    const float* ad2 = (const float*)d_in[10];
    const float* b2  = (const float*)d_in[11];
    const float* W3 = (const float*)d_in[12];
    const float* as3 = (const float*)d_in[13];
    const float* ad3 = (const float*)d_in[14];
    const float* b3  = (const float*)d_in[15];
    float* out = (float*)d_out;

    // workspace carve (all 256-aligned)
    char* w = (char*)d_ws;
    float*    h0     = (float*)   (w + 0);            // 50000*256 f32
    float*    h1     = (float*)   (w + 51200000);     // 50000*256 f32
    float*    als    = (float*)   (w + 102400000);    // 50000*4
    float*    ald    = (float*)   (w + 103200000);    // 50000*4
    unsigned* mmax   = (unsigned*)(w + 104000000);    // 50000*4
    float*    den    = (float*)   (w + 104800000);    // 50000*4
    float*    eex    = (float*)   (w + 105600000);    // 850000*4
    float*    pooled = (float*)   (w + 119200000);    // 512*10

    const int TB = 256;
    int gEdge   = cdiv(E_TOT, TB);
    int gEdge32 = cdiv((long long)E_TOT * 32, TB);
    int numMt   = cdiv(N_NODES, 16);
    int gGemm   = numMt < 1024 ? numMt : 1024;        // amortize W staging

    // post-GEMM edge pipeline for one layer
    auto run_edges = [&](const float* gout, const float* asrc, const float* adst,
                         const float* bias, float* aout, int HC, int H, int act) {
        int C = HC / H;
        attn_coef<<<cdiv(N_NODES * H, TB), TB, 0, stream>>>(gout, asrc, adst, als, ald, N_NODES, H, C);

        fill_u32<<<cdiv(N_NODES * H, TB), TB, 0, stream>>>(mmax, ENC_NEG_INF, N_NODES * H);
        fill_f32<<<cdiv(N_NODES * H, TB), TB, 0, stream>>>(den, 0.0f, N_NODES * H);
        fill_f32<<<cdiv((long long)N_NODES * HC, TB), TB, 0, stream>>>(aout, 0.0f, N_NODES * HC);

        edge_max_kernel<<<gEdge, TB, 0, stream>>>(ei, als, ald, mmax, H);
        edge_exp_kernel<<<gEdge, TB, 0, stream>>>(ei, als, ald, mmax, eex, den, H);
        edge_aggregate<<<gEdge32, TB, 0, stream>>>(ei, gout, eex, den, aout, HC, H);

        bias_act<<<cdiv((long long)N_NODES * HC, TB), TB, 0, stream>>>(aout, bias, N_NODES, HC, SLOPE_ACT, act);
    };

    // layer 1: 128 -> 4x64 (concat 256); KC=4, NT=16
    gemm_wmma<4, 16><<<gGemm, GEMM_THREADS, 0, stream>>>(x, W1, h0, N_NODES, HC1);
    run_edges(h0, as1, ad1, b1, h1, HC1, HEADS1, 1);

    // layer 2: 256 -> 128; KC=8, NT=8
    gemm_wmma<8, 8><<<gGemm, GEMM_THREADS, 0, stream>>>(h1, W2, h0, N_NODES, HID2);
    run_edges(h0, as2, ad2, b2, h1, HID2, 1, 1);

    // layer 3: 128 -> 10 (Npad=16); KC=4, NT=1, no activation
    gemm_wmma<4, 1><<<gGemm, GEMM_THREADS, 0, stream>>>(h1, W3, h0, N_NODES, N_CLASSES);
    run_edges(h0, as3, ad3, b3, h1, N_CLASSES, 1, 0);

    // global add pool + log softmax (tuple output: [log_softmax, pooled])
    fill_f32<<<cdiv(N_GRAPHS * N_CLASSES, TB), TB, 0, stream>>>(pooled, 0.0f, N_GRAPHS * N_CLASSES);
    pool_kernel<<<cdiv(N_NODES * N_CLASSES, TB), TB, 0, stream>>>(h1, batch, pooled);
    lsm_kernel<<<cdiv(N_GRAPHS, TB), TB, 0, stream>>>(pooled, out);
}
